// _CrossNeuronBlock_49091476193304
// MI455X (gfx1250) — matrix operand order repaired and checked by reading, not compiled, not run on hardware
//
#include <hip/hip_runtime.h>

typedef unsigned short ushort_t;
typedef __attribute__((ext_vector_type(16))) __bf16        v16bf;
typedef __attribute__((ext_vector_type(8)))  float         v8f;
typedef __attribute__((ext_vector_type(8)))  unsigned int  v8u;

__device__ __forceinline__ ushort_t f2bf(float f) {
  unsigned int u = __float_as_uint(f);
  u += 0x7FFFu + ((u >> 16) & 1u);            // round-to-nearest-even
  return (ushort_t)(u >> 16);
}
__device__ __forceinline__ float bf2f(ushort_t h) {
  return __uint_as_float(((unsigned int)h) << 16);
}

// ------------------------------------ f32 -> bf16 with zero padding to (Md,Kd)
__global__ void k_wpad(const float* __restrict__ src, ushort_t* __restrict__ dst,
                       int Ms, int Ks, int Md, int Kd) {
  int i = blockIdx.x * 256 + threadIdx.x;
  if (i >= Md * Kd) return;
  int r = i / Kd, k = i - r * Kd;
  float v = (r < Ms && k < Ks) ? src[r * Ks + k] : 0.0f;
  dst[i] = f2bf(v);
}

__global__ void k_bpad(const float* __restrict__ src, float* __restrict__ dst, int ns, int nd) {
  int i = blockIdx.x * 256 + threadIdx.x;
  if (i < nd) dst[i] = (i < ns) ? src[i] : 0.0f;
}

// ------------------------------------------- gather NCHW tiles -> xv[B][S][C]
__global__ void k_gather(const float* __restrict__ x, ushort_t* __restrict__ xv) {
  __shared__ float tile[32][33];
  int tid = threadIdx.x;
  int c0 = blockIdx.x * 32;
  int s0 = blockIdx.y * 32;
  int Bi = blockIdx.z;
  int chunk = Bi >> 4, bt = Bi & 15;
  int ty = (chunk >> 1) * 24, tx = (chunk & 1) * 24;
  int i = tid & 31;            // s (fast) for coalesced x reads
  int jb = tid >> 5;
#pragma unroll
  for (int r = 0; r < 4; ++r) {
    int j = jb + 8 * r;        // c
    int s = s0 + i, c = c0 + j;
    int h = ty + s / 24, w = tx + s % 24;
    tile[i][j] = x[(((long)bt * 1024 + c) * 48 + h) * 48 + w];
  }
  __syncthreads();
  int jj = tid & 31;           // c (fast) for coalesced xv writes
  int ib = tid >> 5;
#pragma unroll
  for (int r = 0; r < 4; ++r) {
    int ii = ib + 8 * r;       // s
    xv[((long)Bi * 576 + (s0 + ii)) * 1024 + (c0 + jj)] = f2bf(tile[ii][jj]);
  }
}

// ---------------------------------------------------- bf16 WMMA GEMM (padded)
// M%32==0 (grid.y=M/32), K%32==0, N%128==0 (grid.x=N/128). 8 waves -> 32x128
// tile, 2 WMMA/K-step per wave. Software-pipelined: next tile's global loads
// are issued before the current tile's WMMA phase. Epilogue fully templated.
template <bool RELU, bool BIAS, bool OUTB, bool OUTF>
__global__ void k_gemm(const ushort_t* __restrict__ A, int lda, long sAb, long sAk,
                       const ushort_t* __restrict__ Bm, int ldb, long sBb, long sBk,
                       ushort_t* __restrict__ Cb, float* __restrict__ Cf,
                       int ldc, long sCb, long sCk,
                       const float* __restrict__ bias,
                       int K, int nblk) {
  __shared__ unsigned int lsa[32 * 16];    // 32 x 32 bf16 (dword = k-pair)
  __shared__ unsigned int lsb[128 * 20];   // [n][k-pair], row stride 20 dwords

  int tid = threadIdx.x;
  int lane = tid & 31;
  int wv = tid >> 5;
  int z = blockIdx.z;
  int b = z / nblk, blk = z % nblk;
  const unsigned int* Ag = (const unsigned int*)(A + (long)b * sAb + (long)blk * sAk);
  const unsigned int* Bg = (const unsigned int*)(Bm + (long)b * sBb + (long)blk * sBk);
  long coff = (long)b * sCb + (long)blk * sCk;

  int m_base = blockIdx.y * 32;
  int n_base = blockIdx.x * 128;
  int ldad = lda >> 1;
  int ldbd = ldb >> 1;

  // staging roles
  int a_row = tid >> 3;                 // 0..31
  int a_dw  = (tid & 7) * 2;            // dword pair within row (16 dwords)
  long a_goff = (long)(m_base + a_row) * ldad + a_dw;
  int kp = tid >> 4;                    // 0..15 k-pair
  int nc = (tid & 15) * 8;              // 8-column chunk
  long b_goff0 = (long)(2 * kp)     * ldbd + ((n_base + nc) >> 1);
  long b_goff1 = (long)(2 * kp + 1) * ldbd + ((n_base + nc) >> 1);

  // pipeline prologue: load first tile into registers
  uint2 va = *(const uint2*)(Ag + a_goff);
  uint4 r0 = *(const uint4*)(Bg + b_goff0);
  uint4 r1 = *(const uint4*)(Bg + b_goff1);

  v8f acc[2] = {v8f{}, v8f{}};
  for (int kb = 0; kb < K; kb += 32) {
    __syncthreads();                    // LDS free (previous compute done)
    lsa[a_row * 16 + a_dw]     = va.x;
    lsa[a_row * 16 + a_dw + 1] = va.y;
    {
      unsigned int q0[4] = {r0.x, r0.y, r0.z, r0.w};
      unsigned int q1[4] = {r1.x, r1.y, r1.z, r1.w};
#pragma unroll
      for (int e = 0; e < 4; ++e) {
        unsigned int lo = (q0[e] & 0xFFFFu) | (q1[e] << 16);          // n even
        unsigned int hi = (q0[e] >> 16)     | (q1[e] & 0xFFFF0000u);  // n odd
        lsb[(nc + 2 * e)     * 20 + kp] = lo;
        lsb[(nc + 2 * e + 1) * 20 + kp] = hi;
      }
    }
    __syncthreads();
    if (kb + 32 < K) {                  // prefetch next tile (uniform branch)
      va = *(const uint2*)(Ag + a_goff + ((kb + 32) >> 1));
      long kadd = (long)(kb + 32) * ldbd;
      r0 = *(const uint4*)(Bg + b_goff0 + kadd);
      r1 = *(const uint4*)(Bg + b_goff1 + kadd);
    }
    // B fragment: two 4-dword runs (table k order) -> 2x ds_load_b128
    v8u bu;
    int nl = wv * 16 + (lane & 15);
    int kh = lane >> 4;
#pragma unroll
    for (int i = 0; i < 8; ++i) bu[i] = lsb[nl * 20 + ((i < 4) ? (kh * 4 + i) : (8 + kh * 4 + (i - 4)))];
    v16bf bv = __builtin_bit_cast(v16bf, bu);
#pragma unroll
    for (int sub = 0; sub < 2; ++sub) {
      v8u au;
      int m = sub * 16 + (lane & 15);
      int kg = lane >> 4;
#pragma unroll
      for (int j = 0; j < 8; ++j) {
        int p = (j < 4) ? (kg * 4 + j) : (8 + kg * 4 + (j - 4));
        au[j] = lsa[m * 16 + p];
      }
      acc[sub] = __builtin_amdgcn_wmma_f32_16x16x32_bf16(
          false, __builtin_bit_cast(v16bf, au),
          false, bv, (short)0, acc[sub], false, false);
    }
  }
  // epilogue (straight-line; D layout: lanes 0-15 N=lane/M=v, 16-31 N=lane-16/M=8+v)
  int col = n_base + wv * 16 + (lane & 15);
#pragma unroll
  for (int sub = 0; sub < 2; ++sub) {
    int rbase = m_base + sub * 16 + 8 * (lane >> 4);
    long idx = coff + (long)rbase * ldc + col;
#pragma unroll
    for (int v = 0; v < 8; ++v) {
      float val = acc[sub][v];
      if (BIAS) val += bias[rbase + v];
      if (RELU) val = fmaxf(val, 0.0f);
      if (OUTB) Cb[idx] = f2bf(val);
      if (OUTF) Cf[idx] = val;
      idx += ldc;
    }
  }
}

// ------------------------------------------------------ channel means over S
__global__ void k_mean(const ushort_t* __restrict__ xv2, float* __restrict__ xm) {
  int c = blockIdx.x * 256 + threadIdx.x;
  int b = blockIdx.y;
  const ushort_t* p = xv2 + (long)b * 576 * 1024 + c;
  float acc = 0.f;
  for (int s = 0; s < 576; ++s) acc += bf2f(p[(long)s * 1024]);
  xm[(long)b * 1024 + c] = acc * (1.0f / 576.0f);
}

// ----------------------------- block-diagonal column softmax -> attn (bf16)
__global__ void k_attn(const float* __restrict__ xm, ushort_t* __restrict__ attn) {
  __shared__ float sx[512];
  int z = blockIdx.x;           // b*2 + blk
  int b = z >> 1, blk = z & 1;
  int tid = threadIdx.x;
  const float* xmb = xm + (long)b * 1024 + blk * 512;
  sx[tid] = xmb[tid];
  sx[tid + 256] = xmb[tid + 256];
  __syncthreads();
  ushort_t* ab = attn + (long)z * 512 * 512;
#pragma unroll
  for (int half = 0; half < 2; ++half) {
    int d = tid + half * 256;
    float xd = sx[d];
    float sum = 0.f;
    for (int cc = 0; cc < 512; ++cc) {
      float diff = xd - sx[cc];
      sum += __expf(-diff * diff);
    }
    float inv = 1.0f / sum;
    for (int cc = 0; cc < 512; ++cc) {
      float diff = xd - sx[cc];
      ab[(long)cc * 512 + d] = f2bf(__expf(-diff * diff) * inv);
    }
  }
}

// -------------------------------------- BatchNorm stats over (B, C) per s
__global__ void k_bnstats(const float* __restrict__ out2,
                          float* __restrict__ mu, float* __restrict__ rsig) {
  __shared__ float s1[256], s2[256];
  int s = blockIdx.x;
  int tid = threadIdx.x;
  float a1 = 0.f, a2 = 0.f;
  for (int b = 0; b < 64; ++b) {
    const float* p = out2 + ((long)b * 576 + s) * 1024;
#pragma unroll
    for (int i = 0; i < 4; ++i) {
      float v = p[tid + 256 * i];
      a1 += v; a2 += v * v;
    }
  }
  s1[tid] = a1; s2[tid] = a2;
  __syncthreads();
  for (int off = 128; off > 0; off >>= 1) {
    if (tid < off) { s1[tid] += s1[tid + off]; s2[tid] += s2[tid + off]; }
    __syncthreads();
  }
  if (tid == 0) {
    float m = s1[0] * (1.0f / 65536.0f);
    float var = s2[0] * (1.0f / 65536.0f) - m * m;
    mu[s] = m;
    rsig[s] = rsqrtf(var + 1e-5f);
  }
}

// --------------- BN apply + scatter-add (x + 0.25*chunk) + ReLU, LDS transpose
__global__ void k_final(const float* __restrict__ x, const float* __restrict__ out2,
                        const float* __restrict__ mu, const float* __restrict__ rsig,
                        const float* __restrict__ gamma, const float* __restrict__ beta,
                        float* __restrict__ out) {
  __shared__ float tile[32][33];
  int tid = threadIdx.x;
  int c0 = blockIdx.x * 32;
  int s0 = blockIdx.y * 32;
  int Bi = blockIdx.z;
  int chunk = Bi >> 4, bt = Bi & 15;
  int ty = (chunk >> 1) * 24, tx = (chunk & 1) * 24;
  int j = tid & 31;            // c (fast) for coalesced out2 reads
  int ibase = tid >> 5;
#pragma unroll
  for (int r = 0; r < 4; ++r) {
    int i = ibase + 8 * r;     // s
    int s = s0 + i;
    float v = out2[((long)Bi * 576 + s) * 1024 + c0 + j];
    tile[i][j] = (v - mu[s]) * rsig[s] * gamma[s] + beta[s];
  }
  __syncthreads();
  int ii = tid & 31;           // s (fast) -> w-contiguous x access
  int jbase = tid >> 5;
#pragma unroll
  for (int r = 0; r < 4; ++r) {
    int jj = jbase + 8 * r;    // c
    int s = s0 + ii, c = c0 + jj;
    int h = ty + s / 24, w = tx + s % 24;
    long xi = (((long)bt * 1024 + c) * 48 + h) * 48 + w;
    out[xi] = fmaxf(x[xi] + 0.25f * tile[ii][jj], 0.0f);
  }
}

extern "C" void kernel_launch(void* const* d_in, const int* in_sizes, int n_in,
                              void* d_out, int out_size, void* d_ws, size_t ws_size,
                              hipStream_t stream) {
  const float* x      = (const float*)d_in[0];
  const float* w_in1  = (const float*)d_in[1];
  const float* b_in1  = (const float*)d_in[2];
  const float* w_in2  = (const float*)d_in[3];
  const float* b_in2  = (const float*)d_in[4];
  const float* w_out1 = (const float*)d_in[5];
  const float* b_out1 = (const float*)d_in[6];
  const float* w_out2 = (const float*)d_in[7];
  const float* b_out2 = (const float*)d_in[8];
  const float* gamma  = (const float*)d_in[9];
  const float* beta   = (const float*)d_in[10];
  float* out = (float*)d_out;

  const long Bb = 64, S = 576, C = 1024, BLK = 512;
  const long Hp = 96;                    // 72 padded to 96 (mult of 32)
  char* ws = (char*)d_ws;
  size_t off = 0;
  auto alloc = [&](size_t bytes) {
    char* p = ws + off;
    off = (off + bytes + 255) & ~(size_t)255;
    return p;
  };

  ushort_t* w1b  = (ushort_t*)alloc(Hp * S * 2);       // 96 x 576, rows>=72 zero
  ushort_t* w2b  = (ushort_t*)alloc(S * Hp * 2);       // 576 x 96, cols>=72 zero
  ushort_t* w3b  = (ushort_t*)alloc(Hp * S * 2);
  ushort_t* w4b  = (ushort_t*)alloc(S * Hp * 2);
  float*    bp1  = (float*)alloc(Hp * 4);
  float*    bp3  = (float*)alloc(Hp * 4);
  ushort_t* xv   = (ushort_t*)alloc(Bb * S * C * 2);   // reused as out1
  ushort_t* hbuf = (ushort_t*)alloc(Bb * Hp * C * 2);  // h1 / h2 (padded rows)
  ushort_t* xv2  = (ushort_t*)alloc(Bb * S * C * 2);
  float*    xm   = (float*)alloc(Bb * C * 4);
  ushort_t* attn = (ushort_t*)alloc(Bb * 2 * BLK * BLK * 2);
  float*    out2 = (float*)alloc(Bb * S * C * 4);
  float*    mu   = (float*)alloc(S * 4);
  float*    rsig = (float*)alloc(S * 4);
  (void)ws_size; (void)in_sizes; (void)n_in; (void)out_size;

  int nwp = (int)((Hp * S + 255) / 256);
  k_wpad<<<nwp, 256, 0, stream>>>(w_in1,  w1b, 72, 576, 96, 576);
  k_wpad<<<nwp, 256, 0, stream>>>(w_in2,  w2b, 576, 72, 576, 96);
  k_wpad<<<nwp, 256, 0, stream>>>(w_out1, w3b, 72, 576, 96, 576);
  k_wpad<<<nwp, 256, 0, stream>>>(w_out2, w4b, 576, 72, 576, 96);
  k_bpad<<<1, 256, 0, stream>>>(b_in1,  bp1, 72, 96);
  k_bpad<<<1, 256, 0, stream>>>(b_out1, bp3, 72, 96);

  k_gather<<<dim3(32, 18, 64), 256, 0, stream>>>(x, xv);

  // h1 = relu(W1p @ xv + b1p)     M=96 K=576 N=1024  (rows 72..95 -> 0)
  k_gemm<true, true, true, false><<<dim3(8, 3, 64), 256, 0, stream>>>(
      w1b, (int)S, 0, 0,
      xv, (int)C, S * C, 0,
      hbuf, nullptr, (int)C, Hp * C, 0,
      bp1, (int)S, 1);
  // xv2 = W2p @ h1 + b2           M=576 K=96 N=1024
  k_gemm<false, true, true, false><<<dim3(8, 18, 64), 256, 0, stream>>>(
      w2b, (int)Hp, 0, 0,
      hbuf, (int)C, Hp * C, 0,
      xv2, nullptr, (int)C, S * C, 0,
      b_in2, (int)Hp, 1);
  k_mean<<<dim3(4, 64), 256, 0, stream>>>(xv2, xm);
  k_attn<<<128, 256, 0, stream>>>(xm, attn);
  // out1 = xv2 @ attn (block-diagonal)  per (b, blk): M=576 K=512 N=512
  k_gemm<false, false, true, false><<<dim3(4, 18, 128), 256, 0, stream>>>(
      xv2, (int)C, S * C, BLK,
      attn, (int)BLK, 2 * BLK * BLK, BLK * BLK,
      xv, nullptr, (int)C, S * C, BLK,
      nullptr, (int)BLK, 2);
  // h2 = relu(W3p @ out1 + b3p)
  k_gemm<true, true, true, false><<<dim3(8, 3, 64), 256, 0, stream>>>(
      w3b, (int)S, 0, 0,
      xv, (int)C, S * C, 0,
      hbuf, nullptr, (int)C, Hp * C, 0,
      bp3, (int)S, 1);
  // out2 = W4p @ h2 + b4 (f32 for BN)
  k_gemm<false, true, false, true><<<dim3(8, 18, 64), 256, 0, stream>>>(
      w4b, (int)Hp, 0, 0,
      hbuf, (int)C, Hp * C, 0,
      nullptr, out2, (int)C, S * C, 0,
      b_out2, (int)Hp, 1);
  k_bnstats<<<576, 256, 0, stream>>>(out2, mu, rsig);
  k_final<<<dim3(32, 18, 64), 256, 0, stream>>>(x, out2, mu, rsig, gamma, beta, out);
}
